// CluTSPSolver_7069516169273
// MI455X (gfx1250) — compile-verified, matrix operand order checked by reading, not run on hardware
//
#include <hip/hip_runtime.h>
#include <hip/hip_bf16.h>

// ---------------------------------------------------------------------------
// Problem constants (from reference): B=256, N=1024, C=100 (+1), D=256, H=16
// ---------------------------------------------------------------------------
#define BB   256
#define NN   1024
#define CC1  101
#define DD   256
#define HH   16
#define QD   16
#define NEGV (-1000000000.0f)

typedef __attribute__((ext_vector_type(16))) __bf16 bf16x16;
typedef __attribute__((ext_vector_type(8)))  float  f32x8;

__device__ __forceinline__ unsigned short f2bf(float f) {
    unsigned int u = __float_as_uint(f);
    unsigned int r = u + 0x7FFFu + ((u >> 16) & 1u);   // round-to-nearest-even
    return (unsigned short)(r >> 16);
}

union FragBF {
    unsigned int u[8];
    bf16x16      v;
};

// ---------------------------------------------------------------------------
// K1: one b128-wide pass over node_embeddings producing BOTH masked partial
// sums. grid = (B, 8 slices), block = 256.
// Thread t: float4 group d4 = t&63, row-group rg = t>>6 (4 rows in flight).
// ---------------------------------------------------------------------------
__global__ void masked_sum_partial(const float* __restrict__ node,
                                   const unsigned char* __restrict__ mask,
                                   const unsigned char* __restrict__ cmask,
                                   float* __restrict__ part,   // [B*8][2*256]
                                   float* __restrict__ cnt)    // [B*8][2]
{
    const int b = blockIdx.x;
    const int s = blockIdx.y;
    const int tid = threadIdx.x;
    const int n0 = s * 128;

    __shared__ unsigned char keep1[128];
    __shared__ unsigned char keep2[128];
    __shared__ float red[8][256];           // [rg | 4+rg][dim]

    if (tid < 128) {
        unsigned char m  = mask [b * NN + n0 + tid];
        unsigned char cm = cmask[b * NN + n0 + tid];
        keep1[tid] = (m == 0);
        keep2[tid] = (m == 0) && (cm == 0);
    }
    __syncthreads();

    const int d4 = tid & 63;
    const int rg = tid >> 6;
    float a1x = 0.f, a1y = 0.f, a1z = 0.f, a1w = 0.f;
    float a2x = 0.f, a2y = 0.f, a2z = 0.f, a2w = 0.f;
    const float* base = node + ((size_t)b * NN + n0) * DD + d4 * 4;
    for (int r = rg; r < 128; r += 4) {
        const float4 v = *(const float4*)(base + (size_t)r * DD);
        if (keep1[r]) { a1x += v.x; a1y += v.y; a1z += v.z; a1w += v.w; }
        if (keep2[r]) { a2x += v.x; a2y += v.y; a2z += v.z; a2w += v.w; }
    }
    red[rg][d4 * 4 + 0] = a1x;  red[rg][d4 * 4 + 1] = a1y;
    red[rg][d4 * 4 + 2] = a1z;  red[rg][d4 * 4 + 3] = a1w;
    red[4 + rg][d4 * 4 + 0] = a2x;  red[4 + rg][d4 * 4 + 1] = a2y;
    red[4 + rg][d4 * 4 + 2] = a2z;  red[4 + rg][d4 * 4 + 3] = a2w;
    __syncthreads();

    const size_t obase = (size_t)(b * 8 + s) * 512;
    const float s1 = red[0][tid % 256] + red[1][tid % 256] + red[2][tid % 256] + red[3][tid % 256];
    const float s2 = red[4][tid % 256] + red[5][tid % 256] + red[6][tid % 256] + red[7][tid % 256];
    part[obase + tid]       = s1;
    part[obase + 256 + tid] = s2;
    if (tid == 0) {
        int c1 = 0, c2 = 0;
        for (int r = 0; r < 128; ++r) { c1 += keep1[r]; c2 += keep2[r]; }
        cnt[(b * 8 + s) * 2 + 0] = (float)c1;
        cnt[(b * 8 + s) * 2 + 1] = (float)c2;
    }
}

// ---------------------------------------------------------------------------
// K2: reduce partials -> uv, uvc; build bf16 context = [uv | current | depot].
// grid = B, block = 256.
// ---------------------------------------------------------------------------
__global__ void reduce_build_context(const float* __restrict__ part,
                                     const float* __restrict__ cnt,
                                     const float* __restrict__ cur,
                                     const float* __restrict__ depot,
                                     float* __restrict__ uvc,            // [B][256] f32
                                     unsigned short* __restrict__ ctxB)  // [B][768] bf16
{
    const int b = blockIdx.x;
    const int d = threadIdx.x;
    float s1 = 0.f, s2 = 0.f, c1 = 0.f, c2 = 0.f;
    for (int s = 0; s < 8; ++s) {
        const size_t base = (size_t)(b * 8 + s) * 512;
        s1 += part[base + d];
        s2 += part[base + 256 + d];
        c1 += cnt[(b * 8 + s) * 2 + 0];
        c2 += cnt[(b * 8 + s) * 2 + 1];
    }
    const float uv = s1 / fmaxf(c1, 1.0f);
    const float uc = s2 / fmaxf(c2, 1.0f);
    uvc[(size_t)b * DD + d] = uc;
    unsigned short* ctx = ctxB + (size_t)b * 768;
    ctx[d]       = f2bf(uv);
    ctx[256 + d] = f2bf(cur  [(size_t)b * DD + d]);
    ctx[512 + d] = f2bf(depot[(size_t)b * DD + d]);
}

// ---------------------------------------------------------------------------
// K3a: flat f32 -> bf16 conversion (packed 2 per thread).
// ---------------------------------------------------------------------------
__global__ void conv_bf16_pack(const float* __restrict__ in,
                               unsigned int* __restrict__ out, int n2)
{
    const int i = blockIdx.x * blockDim.x + threadIdx.x;
    if (i < n2) {
        const float2 v = ((const float2*)in)[i];
        out[i] = (unsigned int)f2bf(v.x) | ((unsigned int)f2bf(v.y) << 16);
    }
}

// ---------------------------------------------------------------------------
// K3b: W[K][N] f32  ->  WT[N][K] bf16 (tiled 32x32 transpose).
// grid = (N/32, K/32), block = (32, 8).
// ---------------------------------------------------------------------------
__global__ void transpose_conv(const float* __restrict__ W,
                               unsigned short* __restrict__ WT,
                               int K, int N)
{
    __shared__ float tile[32][33];
    const int n0 = blockIdx.x * 32, k0 = blockIdx.y * 32;
    const int tx = threadIdx.x, ty = threadIdx.y;
    for (int i = ty; i < 32; i += 8)
        tile[i][tx] = W[(size_t)(k0 + i) * N + n0 + tx];
    __syncthreads();
    for (int i = ty; i < 32; i += 8)
        WT[(size_t)(n0 + i) * K + k0 + tx] = f2bf(tile[tx][i]);
}

// ---------------------------------------------------------------------------
// K4: LDS-free bf16 WMMA GEMM:  C[M,N] = A[M,K] @ BT[N,K]^T   (f32 accumulate)
// A, BT row-major bf16.  Fully K-unrolled: each lane's A/B fragment is two
// aligned b128 loads per 16x16x32 step (matches ISA 7.12.2 VGPR layout).
// block = 256 (8 waves); wave w owns tile (blockIdx.x*16, (blockIdx.y*8+w)*16).
// ---------------------------------------------------------------------------
template <int K>
__global__ void wmma_gemm_bf16(const unsigned short* __restrict__ A,
                               const unsigned short* __restrict__ BT,
                               float* __restrict__ C, int M, int N)
{
    const int lane = threadIdx.x & 31;
    const int wave = threadIdx.x >> 5;
    const int m0   = blockIdx.x * 16;
    const int n0   = (blockIdx.y * 8 + wave) * 16;
    const int row  = lane & 15;
    const int half = lane >> 4;

    const unsigned short* ap = A  + (size_t)(m0 + row) * K + half * 8;
    const unsigned short* bp = BT + (size_t)(n0 + row) * K + half * 8;

    f32x8 acc = {};
#pragma unroll
    for (int k0 = 0; k0 < K; k0 += 32) {
        FragBF fa, fb;
        const uint4 alo = *(const uint4*)(ap + k0);        // K pairs (h*8 .. h*8+7)
        const uint4 ahi = *(const uint4*)(ap + k0 + 16);   // K pairs (16+h*8 ..)
        const uint4 blo = *(const uint4*)(bp + k0);
        const uint4 bhi = *(const uint4*)(bp + k0 + 16);
        fa.u[0] = alo.x; fa.u[1] = alo.y; fa.u[2] = alo.z; fa.u[3] = alo.w;
        fa.u[4] = ahi.x; fa.u[5] = ahi.y; fa.u[6] = ahi.z; fa.u[7] = ahi.w;
        fb.u[0] = blo.x; fb.u[1] = blo.y; fb.u[2] = blo.z; fb.u[3] = blo.w;
        fb.u[4] = bhi.x; fb.u[5] = bhi.y; fb.u[6] = bhi.z; fb.u[7] = bhi.w;
        acc = __builtin_amdgcn_wmma_f32_16x16x32_bf16(
                  /*neg_a=*/false, fa.v, /*neg_b=*/false, fb.v,
                  /*c_mod=*/(short)0, acc, /*reuse_a=*/false, /*reuse_b=*/false);
    }

    // C/D layout: VGPR v -> M = half*8 + v, N = lane&15.
#pragma unroll
    for (int v = 0; v < 8; ++v)
        C[(size_t)(m0 + half * 8 + v) * N + n0 + row] = acc[v];
}

// ---------------------------------------------------------------------------
// K5: per-batch attention, Wo projection, logits, log-softmax, argmax,
// and all four outputs. grid = B, block = 128.
// ---------------------------------------------------------------------------
__global__ void attn_final(const float* __restrict__ q,     // [B][256]
                           const float* __restrict__ kp,    // [B][101][256]
                           const float* __restrict__ vp,
                           const float* __restrict__ pp,
                           const float* __restrict__ clu,   // [B][101][256]
                           const float* __restrict__ uvc,   // [B][256]
                           const float* __restrict__ cur,
                           const float* __restrict__ depot,
                           const unsigned char* __restrict__ visited, // [B][101]
                           const float* __restrict__ Wo,    // [256][256]
                           float* __restrict__ out_aug,     // [B][1024]
                           float* __restrict__ out_gemb,    // [B][256]
                           float* __restrict__ out_gidx,    // [B]
                           float* __restrict__ out_prob)    // [B][101]
{
    const int b = blockIdx.x, tid = threadIdx.x;
    __shared__ float qs[256], sc[HH * 104], gl[256], g2[256], lg[104];
    __shared__ unsigned char vm[104];
    __shared__ int   sel;
    __shared__ float lse;

    for (int i = tid; i < 256; i += 128) qs[i] = q[(size_t)b * 256 + i];
    for (int i = tid; i < CC1; i += 128) vm[i] = visited[b * CC1 + i];
    __syncthreads();
    if (tid == 0) {
        bool all = true;
        for (int c = 1; c < CC1; ++c) if (!vm[c]) { all = false; break; }
        vm[0] = all ? 0 : 1;                 // vm[0] = ~all_real_visited
    }
    __syncthreads();

    // scores (16 heads x 101 keys), scale 1/sqrt(16), masked.
    const float* kb = kp + (size_t)b * CC1 * DD;
    for (int idx = tid; idx < HH * CC1; idx += 128) {
        const int h = idx / CC1, c = idx % CC1;
        const float* kr = kb + (size_t)c * DD + h * QD;
        float dot = 0.f;
#pragma unroll
        for (int e = 0; e < QD; ++e) dot += qs[h * QD + e] * kr[e];
        sc[h * 104 + c] = vm[c] ? NEGV : dot * 0.25f;
    }
    __syncthreads();

    // softmax per head.
    if (tid < HH) {
        const int h = tid;
        float m = -3.402823466e+38f;
        for (int c = 0; c < CC1; ++c) m = fmaxf(m, sc[h * 104 + c]);
        float s = 0.f;
        for (int c = 0; c < CC1; ++c) { float e = __expf(sc[h * 104 + c] - m); sc[h * 104 + c] = e; s += e; }
        const float inv = 1.0f / s;
        for (int c = 0; c < CC1; ++c) sc[h * 104 + c] *= inv;
    }
    __syncthreads();

    // glimpse then Wo projection.
    const float* vb = vp + (size_t)b * CC1 * DD;
    for (int idx = tid; idx < DD; idx += 128) {
        const int h = idx >> 4, e = idx & 15;
        float g = 0.f;
        for (int c = 0; c < CC1; ++c) g += sc[h * 104 + c] * vb[(size_t)c * DD + h * QD + e];
        gl[idx] = g;
    }
    __syncthreads();
    for (int d = tid; d < DD; d += 128) {
        float s = 0.f;
        for (int i = 0; i < DD; ++i) s += gl[i] * Wo[(size_t)i * DD + d];
        g2[d] = s;
    }
    __syncthreads();

    // logits -> tanh clip -> mask.
    const float* pb = pp + (size_t)b * CC1 * DD;
    for (int c = tid; c < CC1; c += 128) {
        float s = 0.f;
        for (int d = 0; d < DD; ++d) s += g2[d] * pb[(size_t)c * DD + d];
        s = tanhf(s * 0.0625f) * 10.0f;      // /sqrt(256), *CLIP
        lg[c] = vm[c] ? NEGV : s;
    }
    __syncthreads();

    if (tid == 0) {
        float m = -3.402823466e+38f; int am = 0;
        for (int c = 0; c < CC1; ++c) if (lg[c] > m) { m = lg[c]; am = c; }
        float s = 0.f;
        for (int c = 0; c < CC1; ++c) s += __expf(lg[c] - m);
        lse = m + __logf(s);
        sel = am;
    }
    __syncthreads();

    for (int c = tid; c < CC1; c += 128)
        out_prob[(size_t)b * CC1 + c] = lg[c] - lse;

    const float* se = clu + ((size_t)b * CC1 + sel) * DD;
    for (int i = tid; i < DD; i += 128) {
        const float sv = se[i];
        out_gemb[(size_t)b * DD + i]         = sv;
        out_aug [(size_t)b * 1024 + i]       = uvc  [(size_t)b * DD + i];
        out_aug [(size_t)b * 1024 + 256 + i] = cur  [(size_t)b * DD + i];
        out_aug [(size_t)b * 1024 + 512 + i] = sv;
        out_aug [(size_t)b * 1024 + 768 + i] = depot[(size_t)b * DD + i];
    }
    if (tid == 0) out_gidx[b] = (float)sel;
}

// ---------------------------------------------------------------------------
// Host side
// ---------------------------------------------------------------------------
extern "C" void kernel_launch(void* const* d_in, const int* in_sizes, int n_in,
                              void* d_out, int out_size, void* d_ws, size_t ws_size,
                              hipStream_t stream) {
    const float* depot = (const float*)d_in[0];   // [B,1,256]
    const float* clu   = (const float*)d_in[1];   // [B,101,256]
    const float* cur   = (const float*)d_in[2];   // [B,1,256]
    const float* node  = (const float*)d_in[3];   // [B,1024,256]
    const unsigned char* mask    = (const unsigned char*)d_in[4];  // bool [B,1,1024]
    const unsigned char* cmask   = (const unsigned char*)d_in[5];  // bool [B,1,1024]
    const unsigned char* visited = (const unsigned char*)d_in[6];  // bool [B,1,101]
    const float* Wq  = (const float*)d_in[7];     // [768,256]
    const float* Wk  = (const float*)d_in[8];     // [256,256]
    const float* Wv  = (const float*)d_in[9];
    const float* Wks = (const float*)d_in[10];
    const float* Wo  = (const float*)d_in[11];

    const size_t MKV = (size_t)BB * CC1;          // 25856 rows
    const size_t SZC = MKV * DD;                  // 6,619,136 elems

    // Workspace carve-up.
    float* ws   = (float*)d_ws;
    float* part = ws;                             // B*8*512 f32
    float* cnt  = part + (size_t)BB * 8 * 512;    // B*16
    float* uvc  = cnt  + (size_t)BB * 16;         // B*256
    float* qv   = uvc  + (size_t)BB * DD;         // B*256
    float* kp   = qv   + (size_t)BB * DD;         // SZC
    float* vp   = kp   + SZC;
    float* pp   = vp   + SZC;
    unsigned short* cluB = (unsigned short*)(pp + SZC);   // SZC bf16
    unsigned short* ctxB = cluB + SZC;            // B*768 bf16
    unsigned short* WqT  = ctxB + (size_t)BB * 768;       // 256*768 bf16
    unsigned short* WkT  = WqT  + (size_t)DD * 768;       // 256*256 bf16
    unsigned short* WvT  = WkT  + (size_t)DD * DD;
    unsigned short* WksT = WvT  + (size_t)DD * DD;

    // Output carve-up (return order: init_aug, guidance_emb, guidance, clu_prob).
    float* out      = (float*)d_out;
    float* out_aug  = out;                          // B*1024
    float* out_gemb = out_aug  + (size_t)BB * 1024; // B*256
    float* out_gidx = out_gemb + (size_t)BB * DD;   // B
    float* out_prob = out_gidx + (size_t)BB;        // B*101

    // 1) single b128 sweep of node_embeddings -> both masked partial sums.
    masked_sum_partial<<<dim3(BB, 8), 256, 0, stream>>>(node, mask, cmask, part, cnt);

    // 2) finish means; build bf16 context rows.
    reduce_build_context<<<BB, 256, 0, stream>>>(part, cnt, cur, depot, uvc, ctxB);

    // 3) one-time conversions: A-operand (cluster emb) and transposed weights.
    {
        const int n2 = (int)(SZC / 2);
        conv_bf16_pack<<<(n2 + 255) / 256, 256, 0, stream>>>(clu, (unsigned int*)cluB, n2);
        transpose_conv<<<dim3(DD / 32, 768 / 32), dim3(32, 8), 0, stream>>>(Wq,  WqT,  768, DD);
        transpose_conv<<<dim3(DD / 32, DD  / 32), dim3(32, 8), 0, stream>>>(Wk,  WkT,  DD,  DD);
        transpose_conv<<<dim3(DD / 32, DD  / 32), dim3(32, 8), 0, stream>>>(Wv,  WvT,  DD,  DD);
        transpose_conv<<<dim3(DD / 32, DD  / 32), dim3(32, 8), 0, stream>>>(Wks, WksT, DD,  DD);
    }

    // 4) LDS-free, fully K-unrolled WMMA GEMMs (bf16 in, f32 acc).
    //    q : (256 x 768) @ (768 x 256)    -> 24 chained v_wmma per wave
    wmma_gemm_bf16<768><<<dim3(BB / 16, DD / 128), 256, 0, stream>>>(ctxB, WqT, qv, BB, DD);
    //    k/v/p : (25856 x 256) @ (256 x 256) -> 8 chained v_wmma per wave
    wmma_gemm_bf16<256><<<dim3((int)(MKV / 16), DD / 128), 256, 0, stream>>>(cluB, WkT,  kp, (int)MKV, DD);
    wmma_gemm_bf16<256><<<dim3((int)(MKV / 16), DD / 128), 256, 0, stream>>>(cluB, WvT,  vp, (int)MKV, DD);
    wmma_gemm_bf16<256><<<dim3((int)(MKV / 16), DD / 128), 256, 0, stream>>>(cluB, WksT, pp, (int)MKV, DD);

    // 5) attention + logits + selection + outputs.
    attn_final<<<BB, 128, 0, stream>>>(qv, kp, vp, pp, clu, uvc, cur, depot,
                                       visited, Wo,
                                       out_aug, out_gemb, out_gidx, out_prob);

    (void)in_sizes; (void)n_in; (void)out_size; (void)ws_size;
}
